// CtmpGIN_38457137168779
// MI455X (gfx1250) — compile-verified, matrix operand order checked by reading, not run on hardware
//
#include <hip/hip_runtime.h>
#include <hip/hip_bf16.h>

typedef _Float16 v16h __attribute__((ext_vector_type(16)));
typedef _Float16 v8h  __attribute__((ext_vector_type(8)));
typedef _Float16 v4h  __attribute__((ext_vector_type(4)));
typedef float    v8f  __attribute__((ext_vector_type(8)));

#define B_        256
#define NUM_COLS_ 512
#define NUM_NODES_ 256
#define VOCAB_    100
#define D_        128
#define N_        (2 * B_ * NUM_NODES_)   /* 131072 */
#define E_INT_    (N_ * 8)                /* 1048576 */
#define MERGED_   (B_ * NUM_NODES_)       /* 65536 */

// ---------------------------------------------------------------- zero fill
__global__ void zero_f4(float4* __restrict__ p, int n4) {
    int t = blockIdx.x * 256 + threadIdx.x;
    if (t < n4) p[t] = make_float4(0.f, 0.f, 0.f, 0.f);
}

// ------------------------------------------------------- embedding gather -> h
// node n = b*512 + c, h[n] = emb[c, x[b,c]]; one wave per node (float4/lane)
__global__ void embed_kernel(const int* __restrict__ x,
                             const float* __restrict__ emb,
                             float* __restrict__ h) {
    unsigned t = blockIdx.x * 256u + threadIdx.x;
    unsigned n = t >> 5, q = t & 31;
    unsigned c = n & 511u, b = n >> 9;
    int xv = x[b * NUM_COLS_ + c];
    const float4* src = (const float4*)(emb + ((size_t)c * VOCAB_ + xv) * D_);
    float4* dstp = (float4*)(h + (size_t)n * D_);
    dstp[q] = src[q];
}

// ------------------------------------------------- stage-1 edge scatter-add
// one wave per edge; 128 f32 atomics (agg fits in 192MB L2)
__global__ void scatter1_kernel(const float* __restrict__ h,
                                const int* __restrict__ src,
                                const int* __restrict__ dst,
                                float* __restrict__ agg) {
    unsigned t = blockIdx.x * 256u + threadIdx.x;
    unsigned e = t >> 5, q = t & 31;
    int s = src[e], d = dst[e];
    float4 v = ((const float4*)(h + (size_t)s * 128))[q];
    float* dp = agg + (size_t)d * 128 + q * 4;
    atomicAdd(dp + 0, v.x); atomicAdd(dp + 1, v.y);
    atomicAdd(dp + 2, v.z); atomicAdd(dp + 3, v.w);
}

// -------------------------------- stage-2 edge projection table (38 x 128)
__global__ void proj_kernel(const float* __restrict__ lt,
                            const float* __restrict__ ew,
                            const float* __restrict__ eb,
                            float* __restrict__ proj) {
    int v = blockIdx.x, c = threadIdx.x;
    float s = eb[c];
#pragma unroll
    for (int k = 0; k < 8; ++k) s += lt[v * 8 + k] * ew[k * 128 + c];
    proj[v * 128 + c] = s;
}

// -------------------------------- stage-2 scatter: relu(h[src]+proj) -> agg
__global__ void scatter2_kernel(const float* __restrict__ h,
                                const int* __restrict__ src,
                                const int* __restrict__ dst,
                                const int* __restrict__ los,
                                const float* __restrict__ proj,
                                float* __restrict__ agg) {
    unsigned t = blockIdx.x * 256u + threadIdx.x;
    unsigned e = t >> 5, q = t & 31;
    int s, d; const float* pv;
    if (e < E_INT_) { s = src[e]; d = dst[e]; pv = proj; }
    else { unsigned j = e - E_INT_; s = (int)j; d = (int)(j + MERGED_);
           pv = proj + (size_t)los[j >> 8] * 128; }
    float4 hv = ((const float4*)(h + (size_t)s * 128))[q];
    float4 p  = ((const float4*)pv)[q];
    float4 m  = make_float4(fmaxf(hv.x + p.x, 0.f), fmaxf(hv.y + p.y, 0.f),
                            fmaxf(hv.z + p.z, 0.f), fmaxf(hv.w + p.w, 0.f));
    float* dp = agg + (size_t)d * 128 + q * 4;
    atomicAdd(dp + 0, m.x); atomicAdd(dp + 1, m.y);
    atomicAdd(dp + 2, m.z); atomicAdd(dp + 3, m.w);
}

// ------------------------------------------------------------- fused MLP
// block = 256 thr (8 waves), 128 rows/block.
// z=(1+eps)h+agg -> [WMMA f16 16x16x32] z@w1+b1 -> LN -> relu -> @w2+b2 -> h
__launch_bounds__(256, 2)
__global__ void mlp_kernel(const float* __restrict__ hin,
                           const float* __restrict__ agg,
                           const float* __restrict__ w1,
                           const float* __restrict__ b1,
                           const float* __restrict__ gam,
                           const float* __restrict__ bet,
                           const float* __restrict__ w2,
                           const float* __restrict__ b2,
                           const float* __restrict__ epsp,
                           float* __restrict__ hout) {
    __shared__ __align__(16) _Float16 wT[128 * 128];  // weights, output-major
    __shared__ __align__(16) _Float16 zA[8][16 * 128]; // per-wave 16x128 tile
    const int tid  = threadIdx.x;
    const int wave = tid >> 5, lane = tid & 31;
    const int n = lane & 15, hi = lane >> 4;
    const int rowbase = blockIdx.x * 128 + wave * 16;
    const float e1 = 1.0f + epsp[0];
    _Float16* zw = zA[wave];

    // w1 -> LDS transposed (wT[outCol][k])
    for (int i = 0; i < 64; ++i) {
        int idx = tid + i * 256;
        int k = idx >> 7, c = idx & 127;
        wT[c * 128 + k] = (_Float16)w1[idx];
    }

    float b1v[8], gv[8], bev[8], b2v[8];
#pragma unroll
    for (int t = 0; t < 8; ++t) {
        int col = t * 16 + n;
        b1v[t] = b1[col]; gv[t] = gam[col]; bev[t] = bet[col]; b2v[t] = b2[col];
    }

    // z = (1+eps)*h + agg  -> f16 tile in LDS (row-major 16x128)
#pragma unroll 4
    for (int m = 0; m < 16; ++m) {
        size_t row = (size_t)(rowbase + m);
        float4 hv = ((const float4*)(hin + row * 128))[lane];
        float4 av = ((const float4*)(agg + row * 128))[lane];
        v4h zh = { (_Float16)(e1 * hv.x + av.x), (_Float16)(e1 * hv.y + av.y),
                   (_Float16)(e1 * hv.z + av.z), (_Float16)(e1 * hv.w + av.w) };
        *(v4h*)(&zw[m * 128 + lane * 4]) = zh;
    }
    __syncthreads();

    // GEMM1: (16x128) @ (128x128), C init = b1 broadcast
    v8f acc[8];
#pragma unroll
    for (int t = 0; t < 8; ++t)
#pragma unroll
        for (int r = 0; r < 8; ++r) acc[t][r] = b1v[t];
#pragma unroll
    for (int kk = 0; kk < 4; ++kk) {
        union { v16h v; v8h h[2]; } afr;
        const _Float16* ap = &zw[n * 128 + kk * 32 + 8 * hi];
        afr.h[0] = *(const v8h*)ap;
        afr.h[1] = *(const v8h*)(ap + 16);
#pragma unroll
        for (int t = 0; t < 8; ++t) {
            union { v16h v; v8h h[2]; } bfr;
            const _Float16* bp = &wT[(t * 16 + n) * 128 + kk * 32 + 8 * hi];
            bfr.h[0] = *(const v8h*)bp;
            bfr.h[1] = *(const v8h*)(bp + 16);
            acc[t] = __builtin_amdgcn_wmma_f32_16x16x32_f16(
                false, afr.v, false, bfr.v, (short)0, acc[t], false, false);
        }
    }

    // LayerNorm (per row, over 128 cols) + relu -> zA (f16)
#pragma unroll
    for (int r = 0; r < 8; ++r) {
        float s1 = 0.f, s2 = 0.f;
#pragma unroll
        for (int t = 0; t < 8; ++t) { float v = acc[t][r]; s1 += v; s2 += v * v; }
        // masks 1,2,4,8 reduce within each 16-lane half (= one row)
        for (int m = 1; m <= 8; m <<= 1) { s1 += __shfl_xor(s1, m); s2 += __shfl_xor(s2, m); }
        float mu  = s1 * (1.0f / 128.0f);
        float var = s2 * (1.0f / 128.0f) - mu * mu;
        float rstd = rsqrtf(var + 1e-5f);
#pragma unroll
        for (int t = 0; t < 8; ++t) {
            float v = (acc[t][r] - mu) * rstd * gv[t] + bev[t];
            v = v > 0.f ? v : 0.f;
            zw[(r + 8 * hi) * 128 + t * 16 + n] = (_Float16)v;
        }
    }
    __syncthreads();

    // w2 -> LDS (reuse wT)
    for (int i = 0; i < 64; ++i) {
        int idx = tid + i * 256;
        int k = idx >> 7, c = idx & 127;
        wT[c * 128 + k] = (_Float16)w2[idx];
    }
    __syncthreads();

    // GEMM2, C init = b2 broadcast
    v8f acc2[8];
#pragma unroll
    for (int t = 0; t < 8; ++t)
#pragma unroll
        for (int r = 0; r < 8; ++r) acc2[t][r] = b2v[t];
#pragma unroll
    for (int kk = 0; kk < 4; ++kk) {
        union { v16h v; v8h h[2]; } afr;
        const _Float16* ap = &zw[n * 128 + kk * 32 + 8 * hi];
        afr.h[0] = *(const v8h*)ap;
        afr.h[1] = *(const v8h*)(ap + 16);
#pragma unroll
        for (int t = 0; t < 8; ++t) {
            union { v16h v; v8h h[2]; } bfr;
            const _Float16* bp = &wT[(t * 16 + n) * 128 + kk * 32 + 8 * hi];
            bfr.h[0] = *(const v8h*)bp;
            bfr.h[1] = *(const v8h*)(bp + 16);
            acc2[t] = __builtin_amdgcn_wmma_f32_16x16x32_f16(
                false, afr.v, false, bfr.v, (short)0, acc2[t], false, false);
        }
    }

    // store new h (C layout: lane=col within tile, vgpr r = row)
#pragma unroll
    for (int t = 0; t < 8; ++t)
#pragma unroll
        for (int r = 0; r < 8; ++r)
            hout[(size_t)(rowbase + r + 8 * hi) * 128 + t * 16 + n] = acc2[t][r];
}

// ---------------------------------------------- per-graph row sum -> output
__global__ void rowsum_kernel(const float* __restrict__ h,
                              float* __restrict__ out, int off) {
    int g = blockIdx.x, d = threadIdx.x;
    float s = 0.f;
    const float* p = h + (size_t)g * NUM_NODES_ * 128 + d;
    for (int j = 0; j < NUM_NODES_; ++j) s += p[j * 128];
    out[g * 256 + off + d] = s;
}

extern "C" void kernel_launch(void* const* d_in, const int* in_sizes, int n_in,
                              void* d_out, int out_size, void* d_ws, size_t ws_size,
                              hipStream_t stream) {
    (void)in_sizes; (void)n_in; (void)out_size; (void)ws_size;
    const int*   x      = (const int*)d_in[0];
    const int*   ei     = (const int*)d_in[1];
    const int*   los    = (const int*)d_in[2];
    const float* emb    = (const float*)d_in[3];
    const float* lt     = (const float*)d_in[4];
    const float* g1_w1  = (const float*)d_in[5];
    const float* g1_b1  = (const float*)d_in[6];
    const float* g1_g   = (const float*)d_in[7];
    const float* g1_be  = (const float*)d_in[8];
    const float* g1_w2  = (const float*)d_in[9];
    const float* g1_b2  = (const float*)d_in[10];
    const float* g1_eps = (const float*)d_in[11];
    const float* g2_w1  = (const float*)d_in[12];
    const float* g2_b1  = (const float*)d_in[13];
    const float* g2_g   = (const float*)d_in[14];
    const float* g2_be  = (const float*)d_in[15];
    const float* g2_w2  = (const float*)d_in[16];
    const float* g2_b2  = (const float*)d_in[17];
    const float* g2_eps = (const float*)d_in[18];
    const float* g2_ew  = (const float*)d_in[19];
    const float* g2_eb  = (const float*)d_in[20];
    float* out = (float*)d_out;

    char* ws = (char*)d_ws;
    float* h    = (float*)ws;                               // N*128 f32 (64MB)
    float* agg  = (float*)(ws + (size_t)N_ * 128 * 4);      // N*128 f32 (64MB)
    float* proj = (float*)(ws + (size_t)N_ * 128 * 4 * 2);  // 38*128 f32

    const int* src  = ei;
    const int* dstn = ei + E_INT_;

    const int zeroBlocks = (N_ * 128 / 4 + 255) / 256;

    embed_kernel<<<(N_ * 32) / 256, 256, 0, stream>>>(x, emb, h);

    for (int i = 0; i < 2; ++i) {
        zero_f4<<<zeroBlocks, 256, 0, stream>>>((float4*)agg, N_ * 128 / 4);
        scatter1_kernel<<<(E_INT_ * 32) / 256, 256, 0, stream>>>(h, src, dstn, agg);
        mlp_kernel<<<N_ / 128, 256, 0, stream>>>(
            h, agg, g1_w1 + (size_t)i * 16384, g1_b1 + i * 128, g1_g + i * 128,
            g1_be + i * 128, g1_w2 + (size_t)i * 16384, g1_b2 + i * 128,
            g1_eps + i, h);
    }
    rowsum_kernel<<<512, 128, 0, stream>>>(h, out, 0);

    const int e2 = E_INT_ + MERGED_;
    for (int i = 0; i < 2; ++i) {
        proj_kernel<<<38, 128, 0, stream>>>(lt, g2_ew + (size_t)i * 8 * 128,
                                            g2_eb + i * 128, proj);
        zero_f4<<<zeroBlocks, 256, 0, stream>>>((float4*)agg, N_ * 128 / 4);
        scatter2_kernel<<<(e2 * 32) / 256, 256, 0, stream>>>(h, src, dstn, los, proj, agg);
        mlp_kernel<<<N_ / 128, 256, 0, stream>>>(
            h, agg, g2_w1 + (size_t)i * 16384, g2_b1 + i * 128, g2_g + i * 128,
            g2_be + i * 128, g2_w2 + (size_t)i * 16384, g2_b2 + i * 128,
            g2_eps + i, h);
    }
    rowsum_kernel<<<512, 128, 0, stream>>>(h, out, 128);
}
